// ComplexLiGRU_Layer_63488206569511
// MI455X (gfx1250) — matrix operand order repaired and checked by reading, not compiled, not run on hardware
//
#include <hip/hip_runtime.h>
#include <math.h>

typedef __attribute__((ext_vector_type(16))) __bf16 v16bf;
typedef __attribute__((ext_vector_type(8)))  float  v8f;

#define BT   19200   // B*T
#define TT   600
#define BB   32
#define EPSF 1e-4f
#define NWG2 64      // phase-2 workgroups

__device__ __forceinline__ v8f wmma_bf16(v16bf a, v16bf b, v8f c) {
  return __builtin_amdgcn_wmma_f32_16x16x32_bf16(false, a, false, b, (short)0, c, false, false);
}

// ---------------- init: zero packed-h ping-pong buffers + step counters ----------------
__global__ __launch_bounds__(256) void init_ws_k(int* hp_i, int* ctr) {
  int i = blockIdx.x * 256 + threadIdx.x;
  if (i < 32768) hp_i[i] = 0;          // 2 x 65536B of packed bf16 h
  if (i < TT) ctr[i] = 0;
}

// ---------------- pack x (fp32 [BT,512]) -> bf16 A-fragment tiles [1200][16][32][16] ---
__global__ __launch_bounds__(256) void pack_x_k(const float* __restrict__ x, __bf16* __restrict__ xp) {
  int gw = (blockIdx.x * 256 + threadIdx.x) >> 5;   // tile id = mt*16 + kt, 19200 tiles
  int lane = threadIdx.x & 31;
  int mt = gw >> 4, kt = gw & 15;
  int row = mt * 16 + (lane & 15);
  int kb = kt * 32 + (lane >> 4) * 8;
  const float* src = x + (size_t)row * 512 + kb;
  __bf16* dst = xp + ((size_t)gw * 32 + lane) * 16;
#pragma unroll
  for (int e = 0; e < 8; ++e) { dst[e] = (__bf16)src[e]; dst[8 + e] = (__bf16)src[16 + e]; }
}

// --------- pack input weights: Wbig[512,2048]=[[Wr,Wi],[-Wi,Wr]] -> B-frag tiles [128][16][32][16]
__global__ __launch_bounds__(256) void pack_W_k(const float* __restrict__ Wr, const float* __restrict__ Wi,
                                                __bf16* __restrict__ wp) {
  int gw = (blockIdx.x * 256 + threadIdx.x) >> 5;   // tile id = nt*16 + kt, 2048 tiles
  int lane = threadIdx.x & 31;
  int nt = gw >> 4, kt = gw & 15;
  int N = nt * 16 + (lane & 15);
  int k0 = kt * 32 + (lane >> 4) * 8;
  int n = N & 1023;
  bool nIm = N >= 1024;
  __bf16* dst = wp + ((size_t)gw * 32 + lane) * 16;
#pragma unroll
  for (int e = 0; e < 16; ++e) {
    int k = k0 + (e & 7) + (e >> 3) * 16;
    float v = (k < 256) ? (nIm ? Wi[(size_t)k * 1024 + n]         :  Wr[(size_t)k * 1024 + n])
                        : (nIm ? Wr[(size_t)(k - 256) * 1024 + n] : -Wi[(size_t)(k - 256) * 1024 + n]);
    dst[e] = (__bf16)v;
  }
}

// --------- pack recurrent weights: Ubig[1024,2048]=[[Wr,Wi],[-Wi,Wr]] -> tiles [128][32][32][16]
__global__ __launch_bounds__(256) void pack_U_k(const float* __restrict__ Wr, const float* __restrict__ Wi,
                                                __bf16* __restrict__ up) {
  int gw = (blockIdx.x * 256 + threadIdx.x) >> 5;   // tile id = nt*32 + kt, 4096 tiles
  int lane = threadIdx.x & 31;
  int nt = gw >> 5, kt = gw & 31;
  int N = nt * 16 + (lane & 15);
  int k0 = kt * 32 + (lane >> 4) * 8;
  int n = N & 1023;
  bool nIm = N >= 1024;
  __bf16* dst = up + ((size_t)gw * 32 + lane) * 16;
#pragma unroll
  for (int e = 0; e < 16; ++e) {
    int k = k0 + (e & 7) + (e >> 3) * 16;
    float v = (k < 512) ? (nIm ? Wi[(size_t)k * 1024 + n]         :  Wr[(size_t)k * 1024 + n])
                        : (nIm ? Wr[(size_t)(k - 512) * 1024 + n] : -Wi[(size_t)(k - 512) * 1024 + n]);
    dst[e] = (__bf16)v;
  }
}

// ---------------- phase 1: GEMM [BT,512]x[512,2048] + fused complex layernorm ----------------
__global__ __launch_bounds__(512) void phase1_k(
    const __bf16* __restrict__ xp, const __bf16* __restrict__ wp,
    const float* __restrict__ g_rr, const float* __restrict__ g_ri, const float* __restrict__ g_ii,
    const float* __restrict__ b_r, const float* __restrict__ b_i,
    float* __restrict__ wbuf) {
  __shared__ float tile[16][2052];          // 16 rows x 2048 (+pad): 131 KB of 320 KB WGP LDS
  int lane = threadIdx.x & 31;
  int wv = threadIdx.x >> 5;                // 16 waves; wave wv owns N range [wv*128, wv*128+128)
  int mt = blockIdx.x;                      // 1200 row-tiles of 16

  v8f acc[8];
#pragma unroll
  for (int n = 0; n < 8; ++n) acc[n] = (v8f){0.f,0.f,0.f,0.f,0.f,0.f,0.f,0.f};

  for (int kt = 0; kt < 16; ++kt) {
    // issue all 9 fragment loads before any WMMA so loadcnt waits stage down
    v16bf a = *(const v16bf*)(xp + (((size_t)mt * 16 + kt) * 32 + lane) * 16);
    v16bf b[8];
#pragma unroll
    for (int n = 0; n < 8; ++n) {
      int nt = wv * 8 + n;
      b[n] = *(const v16bf*)(wp + (((size_t)nt * 16 + kt) * 32 + lane) * 16);
    }
    __builtin_amdgcn_sched_barrier(0);      // keep loads grouped ahead of the WMMAs
#pragma unroll
    for (int n = 0; n < 8; ++n) acc[n] = wmma_bf16(a, b[n], acc[n]);
  }
  // D layout: lane<16 -> rows 0..7, lane>=16 -> rows 8..15; N = lane&15
  int r0 = (lane < 16) ? 0 : 8;
  int nc = lane & 15;
#pragma unroll
  for (int n = 0; n < 8; ++n) {
    int cb = wv * 128 + n * 16 + nc;
#pragma unroll
    for (int q = 0; q < 8; ++q) tile[r0 + q][cb] = acc[n][q];
  }
  __syncthreads();

  // wave wv normalizes row wv (r = cols 0..1023, i = cols 1024..2047)
  float s_r = 0, s_i = 0, s_rr = 0, s_ii = 0, s_ri = 0;
#pragma unroll 4
  for (int q = 0; q < 32; ++q) {
    int c = lane + q * 32;
    float r = tile[wv][c], i = tile[wv][c + 1024];
    s_r += r; s_i += i; s_rr += r * r; s_ii += i * i; s_ri += r * i;
  }
#pragma unroll
  for (int m = 16; m >= 1; m >>= 1) {
    s_r += __shfl_xor(s_r, m, 32); s_i += __shfl_xor(s_i, m, 32);
    s_rr += __shfl_xor(s_rr, m, 32); s_ii += __shfl_xor(s_ii, m, 32);
    s_ri += __shfl_xor(s_ri, m, 32);
  }
  const float invN = 1.f / 1024.f;
  float mr = s_r * invN, mi = s_i * invN;
  float Vrr = s_rr * invN - mr * mr + EPSF;
  float Vii = s_ii * invN - mi * mi + EPSF;
  float Vri = s_ri * invN - mr * mi;
  float s  = sqrtf(Vrr * Vii - Vri * Vri);
  float tt = sqrtf(Vrr + Vii + 2.f * s);
  float inv = 1.f / (s * tt);
  float Wrr = (Vii + s) * inv, Wii = (Vrr + s) * inv, Wri = -Vri * inv;

  float* o = wbuf + ((size_t)mt * 16 + wv) * 2048;
#pragma unroll 4
  for (int q = 0; q < 32; ++q) {
    int c = lane + q * 32;
    float r = tile[wv][c] - mr, i = tile[wv][c + 1024] - mi;
    float nr = Wrr * r + Wri * i, ni = Wri * r + Wii * i;
    o[c]        = g_rr[c] * nr + g_ri[c] * ni + b_r[c];
    o[c + 1024] = g_ri[c] * nr + g_ii[c] * ni + b_i[c];
  }
}

// ------- phase 2: persistent recurrence, 64 WGs x 16 waves, h kept bf16-packed + in-register ----
__global__ __launch_bounds__(512) void phase2_k(
    const __bf16* __restrict__ up, const float* __restrict__ wbuf,
    __bf16* __restrict__ hp, int* __restrict__ ctr, float* __restrict__ out) {
  __shared__ float lds[16][16][20];                    // padded: rows 8 apart land 32 banks apart
  int lane = threadIdx.x & 31, wv = threadIdx.x >> 5;  // 16 waves
  int j = blockIdx.x;                                  // 0..63: owns h cols [16j, 16j+16)
  int tile_id = wv >> 2;                               // 0..3 = is_z*2 + mtw
  int kw = wv & 3;                                     // K quarter
  int mtw = tile_id & 1;
  int is_z = tile_id >> 1;
  int nt = is_z * 64 + j;                              // B-tile column block in Upacked

  // epilogue element owned by this thread: h[em, col]
  int em = threadIdx.x >> 4;                           // 0..31
  int ec = threadIdx.x & 15;                           // 0..15
  int col = j * 16 + ec;
  int r = em & 15, mt2 = em >> 4;
  int ktw = col >> 5, ko = col & 31;
  int lane_p = r + ((ko >> 3) & 1) * 16;               // packed-A lane for this (row, K)
  int elem_p = (ko & 7) + ((ko & 16) ? 8 : 0);         // packed-A element within v16bf
  size_t hp_off = ((size_t)(mt2 * 32 + ktw) * 32 + lane_p) * 16 + elem_p;
  const size_t HPN = 32768;                            // bf16 elems per h buffer

  float h_reg = 0.f;                                   // fp32 state lives in a register

  for (int t = 0; t < TT; ++t) {
    const __bf16* hc = hp + (t & 1) * HPN;
    __bf16* hx = hp + ((t + 1) & 1) * HPN;

    // issue all 16 fragment loads of this step's k-chain before the first WMMA
    v16bf a[8], b[8];
#pragma unroll
    for (int q = 0; q < 8; ++q) {
      int kt = kw * 8 + q;
      a[q] = *(const v16bf*)(hc + (((size_t)mtw * 32 + kt) * 32 + lane) * 16);
      b[q] = *(const v16bf*)(up + (((size_t)nt * 32 + kt) * 32 + lane) * 16);
    }
    __builtin_amdgcn_sched_barrier(0);
    v8f acc = (v8f){0.f,0.f,0.f,0.f,0.f,0.f,0.f,0.f};
#pragma unroll
    for (int q = 0; q < 8; ++q) acc = wmma_bf16(a[q], b[q], acc);

    {
      int r0 = (lane < 16) ? 0 : 8, c = lane & 15;
#pragma unroll
      for (int q = 0; q < 8; ++q) lds[wv][r0 + q][c] = acc[q];
    }
    __syncthreads();

    float av = 0.f, zv = 0.f;
#pragma unroll
    for (int kq = 0; kq < 4; ++kq) {
      av += lds[mt2 * 4 + kq][r][ec];
      zv += lds[8 + mt2 * 4 + kq][r][ec];
    }
    size_t wrow = (size_t)(em * TT + t) * 2048;
    av += wbuf[wrow + col];
    zv += wbuf[wrow + 1024 + col];
    float z = 1.f / (1.f + __expf(-zv));
    float hcand = av > 0.f ? av : 0.f;
    h_reg = z * h_reg + (1.f - z) * hcand;
    hx[hp_off] = (__bf16)h_reg;
    out[(size_t)(em * TT + t) * 1024 + col] = h_reg;

    __threadfence();
    __syncthreads();
    if (threadIdx.x == 0) {
      __hip_atomic_fetch_add(&ctr[t], 1, __ATOMIC_RELEASE, __HIP_MEMORY_SCOPE_AGENT);
      while (__hip_atomic_load(&ctr[t], __ATOMIC_ACQUIRE, __HIP_MEMORY_SCOPE_AGENT) < NWG2)
        __builtin_amdgcn_s_sleep(1);
    }
    __syncthreads();
  }
}

extern "C" void kernel_launch(void* const* d_in, const int* in_sizes, int n_in,
                              void* d_out, int out_size, void* d_ws, size_t ws_size,
                              hipStream_t stream) {
  const float* x    = (const float*)d_in[0];
  const float* Wr_w = (const float*)d_in[1];
  const float* Wi_w = (const float*)d_in[2];
  const float* Wr_u = (const float*)d_in[3];
  const float* Wi_u = (const float*)d_in[4];
  const float* g_rr = (const float*)d_in[5];
  const float* g_ri = (const float*)d_in[6];
  const float* g_ii = (const float*)d_in[7];
  const float* b_r  = (const float*)d_in[8];
  const float* b_i  = (const float*)d_in[9];
  float* out = (float*)d_out;

  char* ws = (char*)d_ws;
  __bf16* xp   = (__bf16*)(ws);                    // 19,660,800 B
  __bf16* wp   = (__bf16*)(ws + 19660800);         //  2,097,152 B
  __bf16* up   = (__bf16*)(ws + 21757952);         //  4,194,304 B
  float*  wbuf = (float*)(ws + 25952256);          // 157,286,400 B
  __bf16* hp   = (__bf16*)(ws + 183238656);        //    131,072 B (2x packed h)
  int*    ctr  = (int*)(ws + 183369728);           //      2,400 B

  init_ws_k<<<256, 256, 0, stream>>>((int*)hp, ctr);
  pack_x_k<<<2400, 256, 0, stream>>>(x, xp);
  pack_W_k<<<256, 256, 0, stream>>>(Wr_w, Wi_w, wp);
  pack_U_k<<<512, 256, 0, stream>>>(Wr_u, Wi_u, up);
  phase1_k<<<1200, 512, 0, stream>>>(xp, wp, g_rr, g_ri, g_ii, b_r, b_i, wbuf);
  phase2_k<<<NWG2, 512, 0, stream>>>(up, wbuf, hp, ctr, out);
}
